// BeamSearch_88605175316856
// MI455X (gfx1250) — matrix operand order repaired and checked by reading, not compiled, not run on hardware
//
#include <hip/hip_runtime.h>
#include <hip/hip_bf16.h>

// MI455X / gfx1250 beam-search video-captioner.
// - bf16 WMMA (v_wmma_f32_16x16x32_bf16) for every GEMM
// - decoder-loop weights (W_out, dec_Wih, dec_Whh, W_cat) pre-converted to
//   bf16 ONCE: halves the dominant ~2.6 GB L2 sweep and removes inner-loop
//   cvt VALU (encoder Wih is read once -> converted inline instead)
// - attention dead-code eliminated (ctx unused in reference)
// - encoder input projections hoisted into one big WMMA GEMM
// - recurrences in single persistent workgroups
// - softmax + top-4 + beam reorder fused into one block per step
// - 8-wave blocks for the per-step GEMMs; global_prefetch two K-tiles ahead

typedef __attribute__((ext_vector_type(16))) __bf16 v16bf;
typedef __attribute__((ext_vector_type(8)))  float  v8f;

#define DEV __device__ __forceinline__

DEV float sigf(float x) { return 1.0f / (1.0f + __expf(-x)); }

DEV v8f wmma_bf16(v16bf a, v16bf b, v8f c) {
  // D(16x16,f32) = A(16x32,bf16) x B(32x16,bf16) + C
  return __builtin_amdgcn_wmma_f32_16x16x32_bf16(false, a, false, b,
                                                 (short)0, c, false, false);
}

// A fragment (16x32 bf16): lane L holds row m = L&15.
// Half-wave K split (matches B): lanes 0-15 -> K {0..15}, lanes 16-31 -> K {16..31}.
DEV v16bf frag_A_f32(const float* rowptr, int k0, int hi, bool valid) {
  v16bf r;
  if (valid) {
    const float* p = rowptr + k0 + hi * 16;
#pragma unroll
    for (int i = 0; i < 16; ++i) { r[i] = (__bf16)p[i]; }
  } else {
#pragma unroll
    for (int i = 0; i < 16; ++i) r[i] = (__bf16)0.0f;
  }
  return r;
}

// B fragment (32x16 bf16): lane L holds column n = L&15; K = k0 + (L>>4)*16 + 0..15.
// Weight stored row-major (N,K): column n's K-run is contiguous in row n.
DEV v16bf frag_B_f32(const float* rowptr, int k0, int hi) {
  v16bf r;
  const float* p = rowptr + k0 + hi * 16;
#pragma unroll
  for (int i = 0; i < 16; ++i) r[i] = (__bf16)p[i];
  return r;
}

DEV v16bf frag_B_bf16(const unsigned short* rowptr, int k0, int hi) {
  v16bf r;
  const __bf16* p = (const __bf16*)(rowptr + k0 + hi * 16);
#pragma unroll
  for (int i = 0; i < 16; ++i) r[i] = p[i];
  return r;
}

// ---------------------------------------------------------------- constants
#define T_LEN   80
#define D_IN    4096
#define HH      512
#define H2      1024
#define G_ENC   2048      // 4*H
#define G_DEC   4096      // 4*H2
#define VOC     32000
#define KB      4         // beams
#define MAXL    20
#define TOKW    21        // MAX_LEN+1

// ---------------------------------------------------------------- kernels

// f32 -> bf16 weight conversion (one-time)
__global__ void k_cvt_bf16(const float* __restrict__ src,
                           unsigned short* __restrict__ dst, int n) {
  __bf16* d = (__bf16*)dst;
  for (int i = blockIdx.x * blockDim.x + threadIdx.x; i < n;
       i += gridDim.x * blockDim.x)
    d[i] = (__bf16)src[i];
}

// Encoder input projections: X[dir][t, j] = sum_k Wih[dir][j,k] * vs[t,k]
// One wave per (16-row, 16-col) tile. grid = (128, 5, 2), block = 32.
__global__ void k_enc_in_gemm(const float* __restrict__ vs,
                              const float* __restrict__ Wf,
                              const float* __restrict__ Wb,
                              float* __restrict__ Xf, float* __restrict__ Xb) {
  const int lane = threadIdx.x;
  const int hi = lane >> 4, lm = lane & 15;
  const float* W = blockIdx.z ? Wb : Wf;
  float* X = blockIdx.z ? Xb : Xf;
  const int m0 = blockIdx.x * 16;            // gate-row tile
  const int n0 = blockIdx.y * 16;            // time tile (80 = 5*16 exact)
  const float* wrow = W + (size_t)(m0 + lm) * D_IN;
  const float* xrow = vs + (size_t)(n0 + lm) * D_IN;
  v8f acc = {};
  for (int k0 = 0; k0 < D_IN; k0 += 32) {
    __builtin_prefetch((const void*)(wrow + k0 + 64), 0, 0);
    acc = wmma_bf16(frag_A_f32(wrow, k0, hi, true), frag_B_f32(xrow, k0, hi), acc);
  }
  const int n = n0 + lm;
#pragma unroll
  for (int r = 0; r < 8; ++r)
    X[(size_t)n * G_ENC + (m0 + r + hi * 8)] = acc[r];
}

// Sequential LSTM recurrence, one block per direction (1024 threads).
// h lives in LDS, c in registers of threads 0..511.
__global__ void k_enc_rnn(const float* __restrict__ Xf, const float* __restrict__ Xb,
                          const float* __restrict__ Whh_f, const float* __restrict__ Whh_b,
                          const float* __restrict__ bf_, const float* __restrict__ bb_,
                          float* __restrict__ video) {
  __shared__ float sh[HH];
  __shared__ float sg[G_ENC];
  const int dir = blockIdx.x;
  const float* X = dir ? Xb : Xf;
  const float* Whh = dir ? Whh_b : Whh_f;
  const float* bias = dir ? bb_ : bf_;
  const int tid = threadIdx.x;
  if (tid < HH) sh[tid] = 0.0f;
  float c = 0.0f;
  __syncthreads();
  for (int s = 0; s < T_LEN; ++s) {
    const int t = dir ? (T_LEN - 1 - s) : s;
#pragma unroll 1
    for (int jj = 0; jj < 2; ++jj) {
      const int j = tid + jj * 1024;
      const float* wr = Whh + (size_t)j * HH;
      float acc = X[(size_t)t * G_ENC + j] + bias[j];
#pragma unroll 4
      for (int u = 0; u < HH; ++u) acc += wr[u] * sh[u];   // sh[u] broadcast
      sg[j] = acc;
    }
    __syncthreads();
    if (tid < HH) {
      const float gi = sg[tid], gf = sg[HH + tid];
      const float gg = sg[2 * HH + tid], go = sg[3 * HH + tid];
      c = sigf(gf) * c + sigf(gi) * tanhf(gg);
      const float h = sigf(go) * tanhf(c);
      sh[tid] = h;
      video[(size_t)t * H2 + dir * HH + tid] = h;
    }
    __syncthreads();
  }
}

// Decoder init: h0 = broadcast(video[-1]), c0 = 0, tokens = BOS, bp = e0
__global__ void k_dec_init(const float* __restrict__ video,
                           float* __restrict__ hcur, float* __restrict__ ccur,
                           int* __restrict__ toks, float* __restrict__ bp) {
  const int tid = threadIdx.x;
  for (int i = tid; i < KB * H2; i += 1024) {
    hcur[i] = video[(size_t)(T_LEN - 1) * H2 + (i & (H2 - 1))];
    ccur[i] = 0.0f;
  }
  if (tid < KB * TOKW) toks[tid] = 1;            // BOS
  if (tid < KB) bp[tid] = (tid == 0) ? 1.0f : 0.0f;
}

// g = x @ dec_Wih.T + h @ dec_Whh.T + b, x = emb[tokens[:,t]] gathered inline.
// Weights pre-converted to bf16. 8 waves/block, one 16-wide N tile per wave.
// grid = 32, block = 256 (covers 4096 gate rows).
__global__ void k_dec_gates(const float* __restrict__ emb,
                            const int* __restrict__ toks, int tcol,
                            const float* __restrict__ hcur,
                            const unsigned short* __restrict__ WihB,
                            const unsigned short* __restrict__ WhhB,
                            const float* __restrict__ b,
                            float* __restrict__ g) {
  const int lane = threadIdx.x & 31, wid = threadIdx.x >> 5;
  const int hi = lane >> 4, lm = lane & 15;
  const int n = (blockIdx.x * 8 + wid) * 16 + lm;
  const bool mval = lm < KB;
  const float* xrow = nullptr;
  if (mval) xrow = emb + (size_t)toks[lm * TOKW + tcol] * H2;
  const float* hrow = hcur + (size_t)lm * H2;
  const unsigned short* wi = WihB + (size_t)n * H2;
  const unsigned short* wh = WhhB + (size_t)n * H2;
  v8f acc = {};
  for (int k0 = 0; k0 < H2; k0 += 32) {
    __builtin_prefetch((const void*)(wi + k0 + 64), 0, 0);
    acc = wmma_bf16(frag_A_f32(xrow, k0, hi, mval), frag_B_bf16(wi, k0, hi), acc);
  }
  for (int k0 = 0; k0 < H2; k0 += 32) {
    __builtin_prefetch((const void*)(wh + k0 + 64), 0, 0);
    acc = wmma_bf16(frag_A_f32(hrow, k0, hi, mval), frag_B_bf16(wh, k0, hi), acc);
  }
  if (hi == 0) {
#pragma unroll
    for (int r = 0; r < KB; ++r)
      g[(size_t)r * G_DEC + n] = acc[r] + b[n];
  }
}

// LSTM cell nonlinearity for 4 beams x 1024 units. grid = 4, block = 1024.
__global__ void k_dec_cell(const float* __restrict__ g, const float* __restrict__ ccur,
                           float* __restrict__ htmp, float* __restrict__ ctmp) {
  const int idx = blockIdx.x * 1024 + threadIdx.x;      // 0..4095
  const int k = idx >> 10, u = idx & (H2 - 1);
  const float* gk = g + (size_t)k * G_DEC;
  const float gi = gk[u], gf = gk[H2 + u], gg = gk[2 * H2 + u], go = gk[3 * H2 + u];
  const float c = sigf(gf) * ccur[idx] + sigf(gi) * tanhf(gg);
  ctmp[idx] = c;
  htmp[idx] = sigf(go) * tanhf(c);
}

// o = tanh([h2 | x] @ W_cat.T + b_cat), W_cat in bf16.
// grid = 8, block = 256 (8 waves -> 64 N tiles total).
__global__ void k_dec_wcat(const float* __restrict__ htmp,
                           const float* __restrict__ emb,
                           const int* __restrict__ toks, int tcol,
                           const unsigned short* __restrict__ WcatB,
                           const float* __restrict__ bcat,
                           float* __restrict__ o) {
  const int lane = threadIdx.x & 31, wid = threadIdx.x >> 5;
  const int hi = lane >> 4, lm = lane & 15;
  const int n = (blockIdx.x * 8 + wid) * 16 + lm;
  const bool mval = lm < KB;
  const float* hrow = htmp + (size_t)lm * H2;
  const float* xrow = nullptr;
  if (mval) xrow = emb + (size_t)toks[lm * TOKW + tcol] * H2;
  const unsigned short* w = WcatB + (size_t)n * (2 * H2);
  v8f acc = {};
  for (int k0 = 0; k0 < H2; k0 += 32)
    acc = wmma_bf16(frag_A_f32(hrow, k0, hi, mval), frag_B_bf16(w, k0, hi), acc);
  for (int k0 = 0; k0 < H2; k0 += 32)
    acc = wmma_bf16(frag_A_f32(xrow, k0, hi, mval), frag_B_bf16(w + H2, k0, hi), acc);
  if (hi == 0) {
#pragma unroll
    for (int r = 0; r < KB; ++r)
      o[(size_t)r * H2 + n] = tanhf(acc[r] + bcat[n]);
  }
}

// logits = o @ W_out.T + b_out, W_out in bf16.
// grid = 250, block = 256 (8 waves -> 2000 N tiles total).
__global__ void k_dec_logits(const float* __restrict__ o,
                             const unsigned short* __restrict__ WoutB,
                             const float* __restrict__ bout,
                             float* __restrict__ logits) {
  const int lane = threadIdx.x & 31, wid = threadIdx.x >> 5;
  const int hi = lane >> 4, lm = lane & 15;
  const int n = (blockIdx.x * 8 + wid) * 16 + lm;
  const bool mval = lm < KB;
  const float* orow = o + (size_t)lm * H2;
  const unsigned short* wrow = WoutB + (size_t)n * H2;
  v8f acc = {};
  for (int k0 = 0; k0 < H2; k0 += 32) {
    __builtin_prefetch((const void*)(wrow + k0 + 64), 0, 0);
    acc = wmma_bf16(frag_A_f32(orow, k0, hi, mval), frag_B_bf16(wrow, k0, hi), acc);
  }
  if (hi == 0) {
#pragma unroll
    for (int r = 0; r < KB; ++r)
      logits[(size_t)r * VOC + n] = acc[r] + bout[n];
  }
}

// Fused: per-row softmax stats + global top-4 over bp*softmax + beam reorder.
// Single block, 1024 threads.
__global__ void k_dec_topk(const float* __restrict__ logits,
                           float* __restrict__ bp, int* __restrict__ toks, int tcol,
                           const float* __restrict__ htmp, const float* __restrict__ ctmp,
                           float* __restrict__ hcur, float* __restrict__ ccur) {
  __shared__ float red[1024];
  __shared__ float rowmax[KB], rowsum[KB];
  __shared__ float sv[1024 * 4];
  __shared__ int   si[1024 * 4];
  __shared__ int   sbi[KB], snt[KB];
  __shared__ float sval[KB];
  const int tid = threadIdx.x;
  float bpl[KB];
#pragma unroll
  for (int r = 0; r < KB; ++r) bpl[r] = bp[r];

  // softmax stats per beam row
  for (int r = 0; r < KB; ++r) {
    float m = -3.4e38f;
    for (int i = tid; i < VOC; i += 1024) m = fmaxf(m, logits[(size_t)r * VOC + i]);
    red[tid] = m; __syncthreads();
    for (int s = 512; s > 0; s >>= 1) {
      if (tid < s) red[tid] = fmaxf(red[tid], red[tid + s]);
      __syncthreads();
    }
    if (tid == 0) rowmax[r] = red[0];
    __syncthreads();
    float sum = 0.0f;
    for (int i = tid; i < VOC; i += 1024)
      sum += __expf(logits[(size_t)r * VOC + i] - rowmax[r]);
    red[tid] = sum; __syncthreads();
    for (int s = 512; s > 0; s >>= 1) {
      if (tid < s) red[tid] += red[tid + s];
      __syncthreads();
    }
    if (tid == 0) rowsum[r] = red[0];
    __syncthreads();
  }

  // local top-4 of bp[r] * softmax over the flattened (KB*VOC) scores
  float tv[4] = {-3.4e38f, -3.4e38f, -3.4e38f, -3.4e38f};
  int   ti[4] = {0, 0, 0, 0};
  for (int i = tid; i < KB * VOC; i += 1024) {
    const int r = i / VOC;
    const float s = bpl[r] * __expf(logits[i] - rowmax[r]) / rowsum[r];
    if (s > tv[3]) {
      tv[3] = s; ti[3] = i;
#pragma unroll
      for (int q = 3; q > 0; --q) {
        if (tv[q] > tv[q - 1]) {
          float f = tv[q]; tv[q] = tv[q - 1]; tv[q - 1] = f;
          int   w = ti[q]; ti[q] = ti[q - 1]; ti[q - 1] = w;
        }
      }
    }
  }
#pragma unroll
  for (int q = 0; q < 4; ++q) { sv[tid * 4 + q] = tv[q]; si[tid * 4 + q] = ti[q]; }

  // tree-merge sorted 4-lists
  for (int s = 512; s > 0; s >>= 1) {
    __syncthreads();
    if (tid < s) {
      float av[4], bv[4]; int ai[4], bi2[4];
#pragma unroll
      for (int q = 0; q < 4; ++q) {
        av[q] = sv[tid * 4 + q];        ai[q]  = si[tid * 4 + q];
        bv[q] = sv[(tid + s) * 4 + q];  bi2[q] = si[(tid + s) * 4 + q];
      }
      int pa = 0, pb = 0;
#pragma unroll
      for (int q = 0; q < 4; ++q) {
        const bool ta = (pb >= 4) || (pa < 4 && av[pa] >= bv[pb]);
        if (ta) { sv[tid * 4 + q] = av[pa]; si[tid * 4 + q] = ai[pa]; ++pa; }
        else    { sv[tid * 4 + q] = bv[pb]; si[tid * 4 + q] = bi2[pb]; ++pb; }
      }
    }
  }
  __syncthreads();

  // beam bookkeeping (tokens + bp) by one thread
  if (tid == 0) {
    int oldtok[KB][TOKW];
    for (int k = 0; k < KB; ++k)
      for (int j = 0; j < TOKW; ++j) oldtok[k][j] = toks[k * TOKW + j];
    for (int q = 0; q < KB; ++q) {
      const int idx = si[q];
      sbi[q] = idx / VOC; snt[q] = idx % VOC; sval[q] = sv[q];
    }
    for (int k = 0; k < KB; ++k) {
      for (int j = 0; j < TOKW; ++j) toks[k * TOKW + j] = oldtok[sbi[k]][j];
      toks[k * TOKW + tcol + 1] = snt[k];
      bp[k] = sval[k];
    }
  }
  __syncthreads();

  // reorder h, c by beam index
  for (int i = tid; i < KB * H2; i += 1024) {
    const int k = i >> 10, u = i & (H2 - 1);
    hcur[i] = htmp[(size_t)sbi[k] * H2 + u];
    ccur[i] = ctmp[(size_t)sbi[k] * H2 + u];
  }
}

// Output: 84 int32 tokens then 4 f32 beam probs, concatenated flat.
__global__ void k_finalize(const int* __restrict__ toks, const float* __restrict__ bp,
                           void* __restrict__ dout) {
  const int tid = threadIdx.x;
  int*   oi = (int*)dout;
  float* of = (float*)dout;
  if (tid < KB * TOKW) oi[tid] = toks[tid];
  if (tid >= KB * TOKW && tid < KB * TOKW + KB) of[tid] = bp[tid - KB * TOKW];
}

// ---------------------------------------------------------------- launcher

extern "C" void kernel_launch(void* const* d_in, const int* in_sizes, int n_in,
                              void* d_out, int out_size, void* d_ws, size_t ws_size,
                              hipStream_t stream) {
  (void)in_sizes; (void)n_in; (void)out_size; (void)ws_size;
  const float* vs    = (const float*)d_in[0];
  const float* emb   = (const float*)d_in[1];
  const float* Wih_f = (const float*)d_in[2];
  const float* Whh_f = (const float*)d_in[3];
  const float* b_f   = (const float*)d_in[4];
  const float* Wih_b = (const float*)d_in[5];
  const float* Whh_b = (const float*)d_in[6];
  const float* b_b   = (const float*)d_in[7];
  const float* dWih  = (const float*)d_in[8];
  const float* dWhh  = (const float*)d_in[9];
  const float* db    = (const float*)d_in[10];
  const float* Wcat  = (const float*)d_in[11];
  const float* bcat  = (const float*)d_in[12];
  const float* Wout  = (const float*)d_in[13];
  const float* bout  = (const float*)d_in[14];

  char* ws = (char*)d_ws;
  size_t off = 0;
  auto alloc = [&](size_t bytes) {
    void* p = ws + off;
    off = (off + bytes + 255) & ~(size_t)255;
    return p;
  };
  unsigned short* WoutB = (unsigned short*)alloc((size_t)VOC * H2 * 2);
  unsigned short* WihB  = (unsigned short*)alloc((size_t)G_DEC * H2 * 2);
  unsigned short* WhhB  = (unsigned short*)alloc((size_t)G_DEC * H2 * 2);
  unsigned short* WcatB = (unsigned short*)alloc((size_t)H2 * (2 * H2) * 2);
  float* Xf    = (float*)alloc((size_t)T_LEN * G_ENC * 4);
  float* Xb    = (float*)alloc((size_t)T_LEN * G_ENC * 4);
  float* video = (float*)alloc((size_t)T_LEN * H2 * 4);
  float* hcur  = (float*)alloc((size_t)KB * H2 * 4);
  float* ccur  = (float*)alloc((size_t)KB * H2 * 4);
  float* htmp  = (float*)alloc((size_t)KB * H2 * 4);
  float* ctmp  = (float*)alloc((size_t)KB * H2 * 4);
  float* gws   = (float*)alloc((size_t)KB * G_DEC * 4);
  float* ows   = (float*)alloc((size_t)KB * H2 * 4);
  float* logit = (float*)alloc((size_t)KB * VOC * 4);
  int*   toks  = (int*)alloc((size_t)KB * TOKW * 4);
  float* bp    = (float*)alloc((size_t)KB * 4);

  // 1) one-time bf16 conversion of all decoder-loop weights
  k_cvt_bf16<<<4096, 256, 0, stream>>>(Wout, WoutB, VOC * H2);
  k_cvt_bf16<<<2048, 256, 0, stream>>>(dWih, WihB, G_DEC * H2);
  k_cvt_bf16<<<2048, 256, 0, stream>>>(dWhh, WhhB, G_DEC * H2);
  k_cvt_bf16<<<1024, 256, 0, stream>>>(Wcat, WcatB, H2 * 2 * H2);

  // 2) encoder input projections as one WMMA GEMM per direction
  k_enc_in_gemm<<<dim3(G_ENC / 16, T_LEN / 16, 2), 32, 0, stream>>>(
      vs, Wih_f, Wih_b, Xf, Xb);

  // 3) sequential recurrences, both directions concurrently
  k_enc_rnn<<<2, 1024, 0, stream>>>(Xf, Xb, Whh_f, Whh_b, b_f, b_b, video);

  // 4) decoder init
  k_dec_init<<<1, 1024, 0, stream>>>(video, hcur, ccur, toks, bp);

  // 5) 20 beam-search steps
  for (int t = 0; t < MAXL; ++t) {
    k_dec_gates<<<G_DEC / 16 / 8, 256, 0, stream>>>(emb, toks, t, hcur, WihB, WhhB, db, gws);
    k_dec_cell<<<4, 1024, 0, stream>>>(gws, ccur, htmp, ctmp);
    k_dec_wcat<<<H2 / 16 / 8, 256, 0, stream>>>(htmp, emb, toks, t, WcatB, bcat, ows);
    k_dec_logits<<<VOC / 16 / 8, 256, 0, stream>>>(ows, WoutB, bout, logit);
    k_dec_topk<<<1, 1024, 0, stream>>>(logit, bp, toks, t, htmp, ctmp, hcur, ccur);
  }

  // 6) write tokens (int32) + bp (f32) flat into d_out
  k_finalize<<<1, 128, 0, stream>>>(toks, bp, d_out);
}
